// GroupedQueryAttention_45097156608165
// MI455X (gfx1250) — matrix operand order repaired and checked by reading, not compiled, not run on hardware
//
#include <hip/hip_runtime.h>

// ---------------- problem constants (match reference) ----------------
#define B_    2
#define T_    2048
#define D_    2048
#define H_    16
#define KV_   4
#define HD_   128
#define NREP_ 4            // H / KV

typedef __bf16 bf16;
typedef __attribute__((ext_vector_type(16))) __bf16 bf16x16;
typedef __attribute__((ext_vector_type(8)))  __bf16 bf16x8;
typedef __attribute__((ext_vector_type(8)))  float  f32x8;

// ---------------- WMMA fragment helpers (CDNA5 wave32 layouts) ----------------
__device__ __forceinline__ bf16x16 concat8(bf16x8 lo, bf16x8 hi) {
  return __builtin_shufflevector(lo, hi, 0,1,2,3,4,5,6,7,8,9,10,11,12,13,14,15);
}

// A fragment: 16(M) x 32(K), row-major source, stride lda (elements).
// ISA 7.12.2: lane<16 -> row=lane, elems 0..7 = K 0..7, elems 8..15 = K 16..23;
//             lane>=16 -> same row, K groups +8.
__device__ __forceinline__ bf16x16 load_a_frag(const bf16* base, int lda) {
  const int lane = threadIdx.x & 31;
  const int row  = lane & 15;
  const int kb   = (lane >> 4) << 3;           // 0 or 8
  const bf16* p  = base + (size_t)row * lda;
  bf16x8 lo = *(const bf16x8*)(p + kb);
  bf16x8 hi = *(const bf16x8*)(p + 16 + kb);
  return concat8(lo, hi);
}

// B fragment: 32(K) x 16(N); source is TRANSPOSED layout Bt[n, k], stride ldb.
// lane<16 -> col=lane, elems 0..7 = K 0..7, elems 8..15 = K 16..23; lane>=16 -> col=lane-16, K groups +8.
__device__ __forceinline__ bf16x16 load_b_frag(const bf16* base, int ldb) {
  const int lane = threadIdx.x & 31;
  const int col  = lane & 15;
  const int kb   = (lane >> 4) << 3;           // 0 or 8
  const bf16* p  = base + (size_t)col * ldb + kb;
  bf16x8 lo = *(const bf16x8*)(p);
  bf16x8 hi = *(const bf16x8*)(p + 16);
  return concat8(lo, hi);
}

__device__ __forceinline__ f32x8 wmma_bf16(bf16x16 a, bf16x16 b, f32x8 c) {
  return __builtin_amdgcn_wmma_f32_16x16x32_bf16(false, a, false, b, (short)0, c, false, false);
}

// ---------------- RoPE tables: cos/sin[t][j], j = 0..63 ----------------
__global__ void rope_table_kernel(float* __restrict__ cosT, float* __restrict__ sinT) {
  int idx = blockIdx.x * blockDim.x + threadIdx.x;
  if (idx >= T_ * 64) return;
  int j = idx & 63;
  int t = idx >> 6;
  // inv_freq = 10000^(-j/64) = exp(-j * ln(10000)/64)
  float inv = __expf(-(float)j * 0.14391156813f);
  float ang = (float)t * inv;
  cosT[idx] = cosf(ang);
  sinT[idx] = sinf(ang);
}

// ---------------- f32 -> bf16 convert ----------------
__global__ void cvt_bf16_kernel(const float* __restrict__ in, bf16* __restrict__ out, int n) {
  int i = blockIdx.x * blockDim.x + threadIdx.x;
  if (i < n) out[i] = (bf16)in[i];
}

// ---------------- f32 (K x N) -> bf16 transposed (N x K) ----------------
__global__ void transpose_cvt_kernel(const float* __restrict__ in, bf16* __restrict__ out,
                                     int K, int N) {
  __shared__ float tile[32][33];
  int k0 = blockIdx.x * 32;
  int n0 = blockIdx.y * 32;
  int tx = threadIdx.x, ty = threadIdx.y;      // block (32, 8)
  #pragma unroll
  for (int r = ty; r < 32; r += 8)
    tile[r][tx] = in[(size_t)(k0 + r) * N + (n0 + tx)];
  __syncthreads();
  #pragma unroll
  for (int r = ty; r < 32; r += 8)
    out[(size_t)(n0 + r) * K + (k0 + tx)] = (bf16)tile[tx][r];
}

// ---------------- QKV projection GEMM: 32x128 tile per wave, fused RoPE ----------------
// MODE 0: Q (rope * 1/sqrt(HD), store [b][h][t][hd]);
// MODE 1: K (rope, store [b][g][t][hd]);
// MODE 2: V (no rope, store transposed [b][g][hd][t])
template <int MODE>
__global__ __launch_bounds__(256, 1)
void proj_kernel(const bf16* __restrict__ xb, const bf16* __restrict__ wt,
                 bf16* __restrict__ out, const float* __restrict__ cosT,
                 const float* __restrict__ sinT, int nheads) {
  const int wave = threadIdx.x >> 5;
  const int wid  = blockIdx.x * (blockDim.x >> 5) + wave;
  const int ROWT = (B_ * T_) / 32;             // 128 row tiles of 32 rows
  const int rowTile = wid % ROWT;
  const int head    = wid / ROWT;
  if (head >= nheads) return;
  const int m0 = rowTile * 32;                 // flat row into (B*T), T%32==0
  const int b  = m0 / T_;
  const int t0 = m0 % T_;

  const bf16* a_base = xb + (size_t)m0 * D_;
  const bf16* b_base = wt + (size_t)head * HD_ * D_;   // Wt is (nheads*HD) x D

  f32x8 acc[2][8] = {};
  for (int k0 = 0; k0 < D_; k0 += 32) {
    bf16x16 a0 = load_a_frag(a_base + k0, D_);
    bf16x16 a1 = load_a_frag(a_base + (size_t)16 * D_ + k0, D_);
    #pragma unroll
    for (int n = 0; n < 8; ++n) {
      bf16x16 bb = load_b_frag(b_base + (size_t)(n * 16) * D_ + k0, D_);
      acc[0][n] = wmma_bf16(a0, bb, acc[0][n]);
      acc[1][n] = wmma_bf16(a1, bb, acc[1][n]);
    }
  }

  const int lane   = threadIdx.x & 31;
  const int col    = lane & 15;
  const int rowoff = (lane >> 4) << 3;
  // Fold the attention 1/sqrt(HD) scale into Q here so the attention hot loop
  // does no per-score scaling.
  const float postscale = (MODE == 0) ? 0.08838834764831845f : 1.0f;

  if (MODE == 2) {
    // V: transposed store Vt[(b*KV + head)*HD + i][t]
    bf16* vbase = out + (size_t)(b * KV_ + head) * HD_ * T_;
    #pragma unroll
    for (int mi = 0; mi < 2; ++mi)
      #pragma unroll
      for (int n = 0; n < 8; ++n) {
        int i = n * 16 + col;
        #pragma unroll
        for (int v = 0; v < 8; ++v) {
          int t = t0 + mi * 16 + v + rowoff;
          vbase[(size_t)i * T_ + t] = (bf16)acc[mi][n][v];
        }
      }
  } else {
    // RoPE: pair (i, i+64) lives in fragments n and n+4 of the same lane.
    bf16* qbase = out + ((size_t)(b * nheads + head) * T_) * HD_;
    #pragma unroll
    for (int mi = 0; mi < 2; ++mi)
      #pragma unroll
      for (int n = 0; n < 4; ++n) {
        int i = n * 16 + col;                  // 0..63 -> freq index j == i
        #pragma unroll
        for (int v = 0; v < 8; ++v) {
          int t   = t0 + mi * 16 + v + rowoff;
          float c = cosT[t * 64 + i] * postscale;
          float s = sinT[t * 64 + i] * postscale;
          float lo = acc[mi][n][v], hi = acc[mi][n + 4][v];
          qbase[(size_t)t * HD_ + i]      = (bf16)(lo * c - hi * s);
          qbase[(size_t)t * HD_ + i + 64] = (bf16)(hi * c + lo * s);
        }
      }
  }
}

// ---------------- one 32-key flash-attention block (MASK only on the diagonal) ----------------
template <bool MASK>
__device__ __forceinline__ void attn_block(
    int s0, int qt, int col, int rowoff,
    const bf16* __restrict__ Kp, const bf16* __restrict__ Vp,
    const bf16x16 (&qf)[4], const bf16x16& ones,
    float (&m_i)[8], float (&l_i)[8], f32x8 (&o)[8], bf16* pb) {
  // ---- S = Q K^T for 32 keys: two 16x16 accumulators (scale pre-folded in Q) ----
  f32x8 sa[2] = {};
  #pragma unroll
  for (int nn = 0; nn < 2; ++nn)
    #pragma unroll
    for (int kk = 0; kk < 4; ++kk) {
      bf16x16 kfrag = load_b_frag(Kp + (size_t)(s0 + nn * 16) * HD_ + kk * 32, HD_);
      sa[nn] = wmma_bf16(qf[kk], kfrag, sa[nn]);
    }

  // ---- (mask +) online-softmax max update per C-fragment row ----
  float alpha_a[8];
  #pragma unroll
  for (int v = 0; v < 8; ++v) {
    float s0v = sa[0][v];
    float s1v = sa[1][v];
    if (MASK) {
      int qi = qt * 16 + v + rowoff;
      if (s0 + col > qi)      s0v = -3.0e38f;
      if (s0 + 16 + col > qi) s1v = -3.0e38f;
    }
    float mv = fmaxf(s0v, s1v);
    #pragma unroll
    for (int off = 1; off < 16; off <<= 1)     // row lives in a 16-lane half
      mv = fmaxf(mv, __shfl_xor(mv, off, 32));
    float mnew = fmaxf(m_i[v], mv);
    alpha_a[v] = __expf(m_i[v] - mnew);
    m_i[v] = mnew;
    float p0 = __expf(s0v - mnew);
    float p1 = __expf(s1v - mnew);
    #pragma unroll
    for (int n = 0; n < 8; ++n) o[n][v] *= alpha_a[v];
    pb[(v + rowoff) * 32 + col]      = (bf16)p0;
    pb[(v + rowoff) * 32 + 16 + col] = (bf16)p1;
  }
  // LDS writes are cross-lane consumed below: wait for DS ops, fence compiler.
  asm volatile("s_wait_dscnt 0" ::: "memory");

  bf16x16 pf = load_a_frag(pb, 32);

  // ---- row sums via matrix pipe: rowsum(P) = P x ones (lands in C-row layout) ----
  f32x8 zero = {};
  f32x8 rs = wmma_bf16(pf, ones, zero);
  #pragma unroll
  for (int v = 0; v < 8; ++v) l_i[v] = l_i[v] * alpha_a[v] + rs[v];

  // ---- O += P V ----
  #pragma unroll
  for (int n = 0; n < 8; ++n) {
    bf16x16 vfrag = load_b_frag(Vp + (size_t)(n * 16) * T_ + s0, T_);
    o[n] = wmma_bf16(pf, vfrag, o[n]);
  }
}

// ---------------- flash attention: one wave = (b, h, 16 query rows) ----------------
__global__ __launch_bounds__(256, 1)
void attn_kernel(const bf16* __restrict__ Qb, const bf16* __restrict__ Kb,
                 const bf16* __restrict__ Vt, bf16* __restrict__ Ob) {
  __shared__ __align__(16) bf16 pbuf[8][16 * 32];   // per-wave P staging (C-layout -> A-layout)
  const int wave = threadIdx.x >> 5;
  const int wid  = blockIdx.x * (blockDim.x >> 5) + wave;
  const int QT   = T_ / 16;
  const int qt = wid % QT;
  const int h  = (wid / QT) % H_;
  const int b  = wid / (QT * H_);
  const int g  = h / NREP_;                    // kv head

  const int col    = (threadIdx.x & 31) & 15;
  const int rowoff = ((threadIdx.x & 31) >> 4) << 3;

  const bf16* Qp = Qb + ((size_t)(b * H_  + h) * T_ + qt * 16) * HD_;
  const bf16* Kp = Kb + ((size_t)(b * KV_ + g) * T_) * HD_;
  const bf16* Vp = Vt + ((size_t)(b * KV_ + g) * HD_) * T_;
  bf16* pb = &pbuf[wave][0];

  bf16x16 qf[4];
  #pragma unroll
  for (int kk = 0; kk < 4; ++kk) qf[kk] = load_a_frag(Qp + kk * 32, HD_);

  bf16x16 ones;
  #pragma unroll
  for (int e = 0; e < 16; ++e) ones[e] = (bf16)1.0f;

  float m_i[8], l_i[8];
  #pragma unroll
  for (int v = 0; v < 8; ++v) { m_i[v] = -3.0e38f; l_i[v] = 0.0f; }
  f32x8 o[8] = {};

  const int kend   = qt * 16 + 16;             // causal key count
  const int dstart = (qt * 16) & ~31;          // first (only) block needing masking

  int s0 = 0;
  for (; s0 < dstart; s0 += 32)                // fully-unmasked main loop
    attn_block<false>(s0, qt, col, rowoff, Kp, Vp, qf, ones, m_i, l_i, o, pb);
  for (; s0 < kend; s0 += 32)                  // single diagonal block
    attn_block<true>(s0, qt, col, rowoff, Kp, Vp, qf, ones, m_i, l_i, o, pb);

  // ---- normalize and store O as (b, t, h*HD) bf16 ----
  bf16* obase = Ob + (size_t)(b * T_ + qt * 16) * (H_ * HD_) + h * HD_;
  #pragma unroll
  for (int v = 0; v < 8; ++v) {
    float inv = 1.0f / l_i[v];
    #pragma unroll
    for (int n = 0; n < 8; ++n)
      obase[(size_t)(v + rowoff) * (H_ * HD_) + n * 16 + col] = (bf16)(o[n][v] * inv);
  }
}

// ---------------- output projection: out = O @ Wo (f32 out), 32x128 tile/wave ----------------
__global__ __launch_bounds__(256, 1)
void oproj_kernel(const bf16* __restrict__ Ob, const bf16* __restrict__ Wot,
                  float* __restrict__ out) {
  const int wave = threadIdx.x >> 5;
  const int wid  = blockIdx.x * (blockDim.x >> 5) + wave;
  const int ROWT = (B_ * T_) / 32;             // 128
  const int rowTile = wid % ROWT;
  const int ntile   = wid / ROWT;              // 0..15, 128 cols each
  const int m0 = rowTile * 32;

  const bf16* a_base = Ob  + (size_t)m0 * D_;
  const bf16* b_base = Wot + (size_t)(ntile * 128) * D_;

  f32x8 acc[2][8] = {};
  for (int k0 = 0; k0 < D_; k0 += 32) {
    bf16x16 a0 = load_a_frag(a_base + k0, D_);
    bf16x16 a1 = load_a_frag(a_base + (size_t)16 * D_ + k0, D_);
    #pragma unroll
    for (int n = 0; n < 8; ++n) {
      bf16x16 bb = load_b_frag(b_base + (size_t)(n * 16) * D_ + k0, D_);
      acc[0][n] = wmma_bf16(a0, bb, acc[0][n]);
      acc[1][n] = wmma_bf16(a1, bb, acc[1][n]);
    }
  }
  const int lane   = threadIdx.x & 31;
  const int col    = lane & 15;
  const int rowoff = (lane >> 4) << 3;
  #pragma unroll
  for (int mi = 0; mi < 2; ++mi)
    #pragma unroll
    for (int n = 0; n < 8; ++n)
      #pragma unroll
      for (int v = 0; v < 8; ++v)
        out[(size_t)(m0 + mi * 16 + v + rowoff) * D_ + ntile * 128 + n * 16 + col] =
            acc[mi][n][v];
}

// ---------------- launch ----------------
extern "C" void kernel_launch(void* const* d_in, const int* in_sizes, int n_in,
                              void* d_out, int out_size, void* d_ws, size_t ws_size,
                              hipStream_t stream) {
  const float* x  = (const float*)d_in[0];
  const float* Wq = (const float*)d_in[1];
  const float* Wk = (const float*)d_in[2];
  const float* Wv = (const float*)d_in[3];
  const float* Wo = (const float*)d_in[4];

  char* base = (char*)d_ws;
  size_t off = 0;
  auto carve = [&](size_t bytes) -> void* {
    void* p = base + off;
    off += (bytes + 255) & ~(size_t)255;
    return p;
  };
  bf16*  xb   = (bf16*)carve((size_t)B_ * T_ * D_ * 2);
  bf16*  Wqt  = (bf16*)carve((size_t)D_ * D_ * 2);
  bf16*  Wkt  = (bf16*)carve((size_t)(KV_ * HD_) * D_ * 2);
  bf16*  Wvt  = (bf16*)carve((size_t)(KV_ * HD_) * D_ * 2);
  bf16*  Wot  = (bf16*)carve((size_t)D_ * D_ * 2);
  bf16*  Qb   = (bf16*)carve((size_t)B_ * H_  * T_ * HD_ * 2);
  bf16*  Kb   = (bf16*)carve((size_t)B_ * KV_ * T_ * HD_ * 2);
  bf16*  Vt   = (bf16*)carve((size_t)B_ * KV_ * HD_ * T_ * 2);
  bf16*  Ob   = (bf16*)carve((size_t)B_ * T_ * D_ * 2);
  float* cosT = (float*)carve((size_t)T_ * 64 * 4);
  float* sinT = (float*)carve((size_t)T_ * 64 * 4);

  rope_table_kernel<<<(T_ * 64) / 256, 256, 0, stream>>>(cosT, sinT);

  { int n = B_ * T_ * D_;
    cvt_bf16_kernel<<<n / 256, 256, 0, stream>>>(x, xb, n); }

  dim3 tb(32, 8);
  transpose_cvt_kernel<<<dim3(D_ / 32, (H_  * HD_) / 32), tb, 0, stream>>>(Wq, Wqt, D_, H_  * HD_);
  transpose_cvt_kernel<<<dim3(D_ / 32, (KV_ * HD_) / 32), tb, 0, stream>>>(Wk, Wkt, D_, KV_ * HD_);
  transpose_cvt_kernel<<<dim3(D_ / 32, (KV_ * HD_) / 32), tb, 0, stream>>>(Wv, Wvt, D_, KV_ * HD_);
  transpose_cvt_kernel<<<dim3(D_ / 32,  D_        / 32), tb, 0, stream>>>(Wo, Wot, D_, D_);

  const int ROWT32 = (B_ * T_) / 32;           // 128
  proj_kernel<0><<<(ROWT32 * H_ ) / 8, 256, 0, stream>>>(xb, Wqt, Qb, cosT, sinT, H_);
  proj_kernel<1><<<(ROWT32 * KV_) / 8, 256, 0, stream>>>(xb, Wkt, Kb, cosT, sinT, KV_);
  proj_kernel<2><<<(ROWT32 * KV_) / 8, 256, 0, stream>>>(xb, Wvt, Vt, cosT, sinT, KV_);

  attn_kernel<<<(B_ * H_ * (T_ / 16)) / 8, 256, 0, stream>>>(Qb, Kb, Vt, Ob);

  oproj_kernel<<<(ROWT32 * 16) / 8, 256, 0, stream>>>(Ob, Wot, (float*)d_out);
}
